// Attention_20555713479370
// MI455X (gfx1250) — matrix operand order repaired and checked by reading, not compiled, not run on hardware
//
#include <hip/hip_runtime.h>
#include <hip/hip_bf16.h>
#include <math.h>

// ---------------------------------------------------------------------------
// Masked multi-head attention for MI455X (gfx1250, wave32, WMMA).
//   q:(32,1024,512)f32  K,V:(8,32,1024,64)f32  mask:(32,1024,1024)bool(1B)
//   W_query:(8,512,64)f32  W_out:(8,64,512)f32  ->  out:(32,1024,512)f32
// Pipeline:
//   1) cvtkv : K -> f16 (row-major), V -> f16 transposed (v-major)   [ws]
//   2) qproj : Qh = NORM * (q @ Wq[h]) as f16, WMMA                   [ws]
//   3) attn  : fused flash attention, double-buffered async-to-LDS
//              tile copies (GLOBAL_LOAD_ASYNC_TO_LDS_B128 + ASYNCcnt),
//              f16 WMMA for QK^T and P.V, online masked softmax
//   4) outproj: heads(32768x512 f16) @ Wo(512x512) -> out f32, WMMA
// Workspace: Qh 32MiB + heads 32MiB + Kh 32MiB + Vth 32MiB = 128MiB.
// ---------------------------------------------------------------------------

#define N_HEADS   8
#define INPUT_DIM 512
#define EMBED_DIM 512
#define KEY_DIM   64
#define BATCH     32
#define N_QUERY   1024
#define GRAPH     1024
#define NORMF     0.125f   // 1/sqrt(64)
#define KPAD      (KEY_DIM + 8)   // padded LDS row (144B, 16B-aligned rows)

typedef __attribute__((ext_vector_type(16))) _Float16 v16h;
typedef __attribute__((ext_vector_type(8)))  _Float16 v8h;
typedef __attribute__((ext_vector_type(4)))  _Float16 v4h;
typedef __attribute__((ext_vector_type(8)))  float    v8f;
typedef __attribute__((ext_vector_type(4)))  int      v4i;

typedef __attribute__((address_space(1))) v4i* gv4i_p;   // global int4*
typedef __attribute__((address_space(3))) v4i* lv4i_p;   // LDS int4*

__device__ __forceinline__ v16h cat8(v8h lo, v8h hi) {
  v16h r;
#pragma unroll
  for (int i = 0; i < 8; ++i) { r[i] = lo[i]; r[i + 8] = hi[i]; }
  return r;
}

__device__ __forceinline__ v8f wmma16(v16h a, v16h b, v8f c) {
  // D(f32 16x16) = A(f16 16x32) * B(f16 32x16) + C
  return __builtin_amdgcn_wmma_f32_16x16x32_f16(false, a, false, b,
                                                (short)0, c, false, false);
}

// --- gfx1250 async global->LDS copy + ASYNCcnt wait ------------------------
#if defined(__has_builtin)
#if __has_builtin(__builtin_amdgcn_global_load_async_to_lds_b128)
#define HAVE_ASYNC_B128 1
#endif
#if __has_builtin(__builtin_amdgcn_s_wait_asynccnt)
#define HAVE_WAIT_ASYNC 1
#endif
#endif

__device__ __forceinline__ void async_copy_b128(_Float16* lds_dst,
                                                const _Float16* gsrc) {
#if defined(HAVE_ASYNC_B128)
  void* g = (void*)gsrc;       // drop const, then addrspacecast to typed ptrs
  void* l = (void*)lds_dst;
  __builtin_amdgcn_global_load_async_to_lds_b128((gv4i_p)g, (lv4i_p)l, 0, 0);
#else
  asm volatile("global_load_async_to_lds_b128 %0, %1, off"
               :: "v"((unsigned)(size_t)lds_dst), "v"(gsrc) : "memory");
#endif
}

#if defined(HAVE_WAIT_ASYNC)
#define WAIT_ASYNC(n) __builtin_amdgcn_s_wait_asynccnt(n)
#else
#define WAIT_ASYNC(n) asm volatile("s_wait_asynccnt " #n ::: "memory")
#endif

// A-fragment (16x32 f16): lane L holds row M=L&15; element e maps to
// k = base8 + (e<8 ? e : e+8), base8=(L>=16)?8:0  -> two contiguous v8h.
// B-fragment (32x16 f16): lane L holds col N=L&15; element e maps to
// k = base8 + (e<8 ? e : e+8), same as A -> two contiguous v8h.

// ---------------------------------------------------------------------------
// Kernel 1: K -> f16 and V -> f16 transposed (per (h,b): Vth[v][g]).
// grid = (8 g-tiles of 128, 256 hb), block = 256.
// ---------------------------------------------------------------------------
__global__ void __launch_bounds__(256) cvtkv_kernel(
    const float* __restrict__ K, const float* __restrict__ V,
    _Float16* __restrict__ Kh, _Float16* __restrict__ Vth) {
  const int hb = blockIdx.y;
  const int g0 = blockIdx.x * 128;
  const int tid = threadIdx.x;
  __shared__ __align__(16) _Float16 T[KEY_DIM][128 + 8];

  const float* Kb = K + (size_t)hb * GRAPH * KEY_DIM;
  const float* Vb = V + (size_t)hb * GRAPH * KEY_DIM;
  _Float16* Ko = Kh + (size_t)hb * GRAPH * KEY_DIM;
  _Float16* Vo = Vth + (size_t)hb * KEY_DIM * GRAPH;

  for (int idx = tid; idx < 128 * 16; idx += 256) {   // K: straight convert
    int g = idx >> 4, kq = (idx & 15) * 4;
    const float* s = Kb + (size_t)(g0 + g) * KEY_DIM + kq;
    v4h d;
#pragma unroll
    for (int j = 0; j < 4; ++j) d[j] = (_Float16)s[j];
    *(v4h*)(Ko + (size_t)(g0 + g) * KEY_DIM + kq) = d;
  }
  for (int idx = tid; idx < 128 * 16; idx += 256) {   // V: transpose via LDS
    int g = idx >> 4, vq = (idx & 15) * 4;
    const float* s = Vb + (size_t)(g0 + g) * KEY_DIM + vq;
#pragma unroll
    for (int j = 0; j < 4; ++j) T[vq + j][g] = (_Float16)s[j];
  }
  __syncthreads();
  for (int idx = tid; idx < 64 * 16; idx += 256) {    // coalesced v-major out
    int v = idx >> 4, gq = (idx & 15) * 8;
    *(v8h*)(Vo + (size_t)v * GRAPH + g0 + gq) = *(const v8h*)&T[v][gq];
  }
}

// ---------------------------------------------------------------------------
// Kernel 2: Q projection.  Qh[h,row,k] = NORM * (q[row,:] @ Wq[h][:,k]), f16.
// block = 256 thr (8 waves x 16 rows = 128 rows), grid = (256 tiles, 8 h).
// ---------------------------------------------------------------------------
__global__ void __launch_bounds__(256) qproj_kernel(
    const float* __restrict__ q, const float* __restrict__ Wq,
    _Float16* __restrict__ Qh) {
  const int h = blockIdx.y;
  const int q0 = blockIdx.x * 128;
  const int tid = threadIdx.x;
  const int wave = tid >> 5, lane = tid & 31;
  const int l16 = lane & 15, hig = lane >> 4;

  __shared__ __align__(16) _Float16 Wt[64][32 + 8];   // Wt[n][k] chunk

  v8f acc[4] = {};
  const float* qrow = q + (size_t)(q0 + wave * 16 + l16) * INPUT_DIM;

  for (int k0 = 0; k0 < INPUT_DIM; k0 += 32) {
    __syncthreads();
    for (int idx = tid; idx < 32 * 64; idx += 256) {
      int kk = idx >> 6, nn = idx & 63;
      Wt[nn][kk] = (_Float16)Wq[((size_t)h * INPUT_DIM + k0 + kk) * KEY_DIM + nn];
    }
    __syncthreads();
    v8h lo, hi8;
    const float* p = qrow + k0 + hig * 8;
#pragma unroll
    for (int i = 0; i < 8; ++i) lo[i]  = (_Float16)p[i];
#pragma unroll
    for (int i = 0; i < 8; ++i) hi8[i] = (_Float16)p[16 + i];
    v16h a = cat8(lo, hi8);
#pragma unroll
    for (int t = 0; t < 4; ++t) {
      const _Float16* bp = &Wt[t * 16 + l16][hig * 8];
      acc[t] = wmma16(a, cat8(*(const v8h*)bp, *(const v8h*)(bp + 16)), acc[t]);
    }
  }
#pragma unroll
  for (int t = 0; t < 4; ++t)
#pragma unroll
    for (int r = 0; r < 8; ++r) {
      int m = q0 + wave * 16 + hig * 8 + r;
      Qh[((size_t)h * (BATCH * N_QUERY) + m) * KEY_DIM + t * 16 + l16] =
          (_Float16)(acc[t][r] * NORMF);
    }
}

// ---------------------------------------------------------------------------
// Kernel 3: fused masked flash attention, Gc=64 chunks, double-buffered
// async tile copies.  grid = (8 q-tiles, 32 b, 8 h), block = 256 (8 waves).
// ---------------------------------------------------------------------------
__device__ __forceinline__ void stage_tile(_Float16 (*Ks)[KPAD],
                                           _Float16 (*Vs)[KPAD],
                                           const _Float16* Kb,
                                           const _Float16* Vb, int g0,
                                           int tid) {
  // 64 rows x 128B each = 512 B128 transfers per operand; 2+2 per thread.
#pragma unroll
  for (int it = 0; it < 2; ++it) {
    int idx = tid + it * 256;
    int n = idx >> 3, part = idx & 7;
    async_copy_b128(&Ks[n][part * 8], Kb + (size_t)(g0 + n) * KEY_DIM + part * 8);
  }
#pragma unroll
  for (int it = 0; it < 2; ++it) {
    int idx = tid + it * 256;
    int v = idx >> 3, part = idx & 7;
    async_copy_b128(&Vs[v][part * 8], Vb + (size_t)v * GRAPH + g0 + part * 8);
  }
}

__global__ void __launch_bounds__(256) attn_kernel(
    const _Float16* __restrict__ Qh, const _Float16* __restrict__ Kh,
    const _Float16* __restrict__ Vth, const unsigned char* __restrict__ mask,
    _Float16* __restrict__ heads) {
  const int h = blockIdx.z, b = blockIdx.y;
  const int q0 = blockIdx.x * 128;
  const int tid = threadIdx.x;
  const int wave = tid >> 5, lane = tid & 31;
  const int l16 = lane & 15, hig = lane >> 4;

  __shared__ __align__(16) _Float16 Ksh[2][KEY_DIM][KPAD];  // [buf][key n][k]
  __shared__ __align__(16) _Float16 Vsh[2][KEY_DIM][KPAD];  // [buf][v][key g]
  __shared__ __align__(16) _Float16 Psh[8][16][KPAD];       // per-wave P

  const size_t hb = (size_t)h * BATCH + b;
  const _Float16* Kb = Kh + hb * GRAPH * KEY_DIM;
  const _Float16* Vb = Vth + hb * KEY_DIM * GRAPH;

  // Q A-fragments for this wave's 16 rows (k=0..31 and k=32..63).
  const _Float16* qp =
      Qh + ((size_t)h * (BATCH * N_QUERY) + (size_t)b * N_QUERY + q0 +
            wave * 16 + l16) * KEY_DIM;
  v16h qa[2];
#pragma unroll
  for (int f = 0; f < 2; ++f) {
    const _Float16* p = qp + f * 32 + hig * 8;
    qa[f] = cat8(*(const v8h*)p, *(const v8h*)(p + 16));
  }

  float mrun[8], lrun[8];
#pragma unroll
  for (int r = 0; r < 8; ++r) { mrun[r] = -__builtin_inff(); lrun[r] = 0.f; }
  v8f o[4] = {};

  const unsigned char* mbase =
      mask + ((size_t)b * N_QUERY + q0 + wave * 16 + hig * 8) * (size_t)GRAPH;

  stage_tile(Ksh[0], Vsh[0], Kb, Vb, 0, tid);   // preload tile 0
  int ibuf = 0;

  for (int g0 = 0; g0 < GRAPH; g0 += 64) {
    const bool hasNext = (g0 + 64) < GRAPH;
    if (hasNext)                                   // overlap copy with math
      stage_tile(Ksh[ibuf ^ 1], Vsh[ibuf ^ 1], Kb, Vb, g0 + 64, tid);
    if (hasNext) { WAIT_ASYNC(4); } else { WAIT_ASYNC(0); }
    __syncthreads();                               // tile ibuf ready for all

    // ---- S = Q K^T (4 n-tiles x 2 k-frags = 8 WMMAs) ----
    v8f s[4] = {};
#pragma unroll
    for (int t = 0; t < 4; ++t) {
      const _Float16* b0 = &Ksh[ibuf][t * 16 + l16][hig * 8];
      const _Float16* b1 = &Ksh[ibuf][t * 16 + l16][32 + hig * 8];
      s[t] = wmma16(qa[0], cat8(*(const v8h*)b0, *(const v8h*)(b0 + 16)), s[t]);
      s[t] = wmma16(qa[1], cat8(*(const v8h*)b1, *(const v8h*)(b1 + 16)), s[t]);
    }

    // ---- masked online softmax over 64 columns ----
#pragma unroll
    for (int r = 0; r < 8; ++r) {
      const unsigned char* mr = mbase + (size_t)r * GRAPH + g0;
      unsigned char mm[4];
      float x[4];
#pragma unroll
      for (int t = 0; t < 4; ++t) {
        mm[t] = mr[t * 16 + l16];
        x[t] = mm[t] ? -__builtin_inff() : s[t][r];
      }
      float tmax = fmaxf(fmaxf(x[0], x[1]), fmaxf(x[2], x[3]));
#pragma unroll
      for (int d = 1; d < 16; d <<= 1) tmax = fmaxf(tmax, __shfl_xor(tmax, d, 32));
      float mnew  = fmaxf(mrun[r], tmax);
      float alpha = (mrun[r] == -__builtin_inff()) ? 0.f : __expf(mrun[r] - mnew);
      float rs = 0.f;
#pragma unroll
      for (int t = 0; t < 4; ++t) {
        float p = mm[t] ? 0.f : __expf(s[t][r] - mnew);
        rs += p;
        Psh[wave][hig * 8 + r][t * 16 + l16] = (_Float16)p;
      }
#pragma unroll
      for (int d = 1; d < 16; d <<= 1) rs += __shfl_xor(rs, d, 32);
      lrun[r] = lrun[r] * alpha + rs;
      mrun[r] = mnew;
#pragma unroll
      for (int t = 0; t < 4; ++t) o[t][r] *= alpha;
    }

    // ---- P (C layout) -> A frags via wave-private LDS; O += P.V ----
    const _Float16* pp = &Psh[wave][l16][0];
    v16h pa0 = cat8(*(const v8h*)(pp + hig * 8), *(const v8h*)(pp + hig * 8 + 16));
    v16h pa1 = cat8(*(const v8h*)(pp + 32 + hig * 8),
                    *(const v8h*)(pp + 32 + hig * 8 + 16));
#pragma unroll
    for (int t = 0; t < 4; ++t) {
      const _Float16* bv = &Vsh[ibuf][t * 16 + l16][0];
      o[t] = wmma16(pa0, cat8(*(const v8h*)(bv + hig * 8),
                              *(const v8h*)(bv + hig * 8 + 16)), o[t]);
      o[t] = wmma16(pa1, cat8(*(const v8h*)(bv + 32 + hig * 8),
                              *(const v8h*)(bv + 32 + hig * 8 + 16)), o[t]);
    }

    __syncthreads();   // all waves done with buf before its next refill
    ibuf ^= 1;
  }

  // Finalize: O / l (0 if fully masked row), heads layout (b, q, h*64+v).
#pragma unroll
  for (int r = 0; r < 8; ++r) {
    float inv = lrun[r] > 0.f ? 1.0f / lrun[r] : 0.0f;
    int m = q0 + wave * 16 + hig * 8 + r;
    _Float16* dst = heads + ((size_t)b * N_QUERY + m) * EMBED_DIM + h * KEY_DIM;
#pragma unroll
    for (int t = 0; t < 4; ++t) dst[t * 16 + l16] = (_Float16)(o[t][r] * inv);
  }
}

// ---------------------------------------------------------------------------
// Kernel 4: output projection. out[row,e] = heads[row,:] @ Wo_flat[:,e].
// grid = (512 row tiles x 4 col tiles), block = 256 (4x2 wave grid).
// ---------------------------------------------------------------------------
__global__ void __launch_bounds__(256) outproj_kernel(
    const _Float16* __restrict__ heads, const float* __restrict__ Wo,
    float* __restrict__ out) {
  const int r0 = blockIdx.x * 64;
  const int e0 = blockIdx.y * 128;
  const int tid = threadIdx.x;
  const int wave = tid >> 5, lane = tid & 31;
  const int l16 = lane & 15, hig = lane >> 4;
  const int wr = wave & 3, wc = wave >> 2;

  __shared__ __align__(16) _Float16 Wot[128][32 + 8];   // [e][k] chunk

  v8f acc[4] = {};
  const _Float16* arp = heads + (size_t)(r0 + wr * 16 + l16) * EMBED_DIM;

  for (int k0 = 0; k0 < EMBED_DIM; k0 += 32) {
    __syncthreads();
    for (int idx = tid; idx < 32 * 32; idx += 256) {
      int kk = idx >> 5, eq = (idx & 31) * 4;
      const float* s = Wo + (size_t)(k0 + kk) * EMBED_DIM + e0 + eq;
#pragma unroll
      for (int j = 0; j < 4; ++j) Wot[eq + j][kk] = (_Float16)s[j];
    }
    __syncthreads();
    const _Float16* p = arp + k0 + hig * 8;
    v16h a = cat8(*(const v8h*)p, *(const v8h*)(p + 16));
#pragma unroll
    for (int t = 0; t < 4; ++t) {
      const _Float16* bp = &Wot[wc * 64 + t * 16 + l16][hig * 8];
      acc[t] = wmma16(a, cat8(*(const v8h*)bp, *(const v8h*)(bp + 16)), acc[t]);
    }
  }
#pragma unroll
  for (int t = 0; t < 4; ++t)
#pragma unroll
    for (int r = 0; r < 8; ++r) {
      int m = r0 + wr * 16 + hig * 8 + r;
      int e = e0 + wc * 64 + t * 16 + l16;
      out[(size_t)m * EMBED_DIM + e] = acc[t][r];
    }
}

// ---------------------------------------------------------------------------
extern "C" void kernel_launch(void* const* d_in, const int* in_sizes, int n_in,
                              void* d_out, int out_size, void* d_ws,
                              size_t ws_size, hipStream_t stream) {
  (void)in_sizes; (void)n_in; (void)out_size; (void)ws_size;
  const float*         q    = (const float*)d_in[0];
  const float*         K    = (const float*)d_in[1];
  const float*         V    = (const float*)d_in[2];
  const unsigned char* mask = (const unsigned char*)d_in[3];  // jnp.bool_: 1B
  const float*         Wq   = (const float*)d_in[4];
  const float*         Wo   = (const float*)d_in[5];
  float*               out  = (float*)d_out;

  const size_t HKV = (size_t)N_HEADS * BATCH * N_QUERY * KEY_DIM;  // 16.78M
  _Float16* Qh    = (_Float16*)d_ws;             // 32 MiB
  _Float16* heads = Qh + HKV;                    // 32 MiB
  _Float16* Kh    = heads + HKV;                 // 32 MiB
  _Float16* Vth   = Kh + HKV;                    // 32 MiB  (128 MiB total)

  cvtkv_kernel<<<dim3(GRAPH / 128, N_HEADS * BATCH), 256, 0, stream>>>(K, V, Kh, Vth);
  qproj_kernel<<<dim3(BATCH * N_QUERY / 128, N_HEADS), 256, 0, stream>>>(q, Wq, Qh);
  attn_kernel<<<dim3(N_QUERY / 128, BATCH, N_HEADS), 256, 0, stream>>>(Qh, Kh, Vth, mask, heads);
  outproj_kernel<<<dim3(BATCH * N_QUERY / 64, EMBED_DIM / 128), 256, 0, stream>>>(heads, Wo, out);
}
